// PathNet_37958920962523
// MI455X (gfx1250) — compile-verified, matrix operand order checked by reading, not run on hardware
//
#include <hip/hip_runtime.h>
#include <hip/hip_bf16.h>

typedef __attribute__((ext_vector_type(16))) _Float16 v16h;
typedef __attribute__((ext_vector_type(8)))  float    v8f;
typedef __attribute__((ext_vector_type(4)))  float    v4f;

#define BATCH   65536
#define IN_DIM  784
#define HW      20      // module width
#define NCLS    10
#define NSTACK  40      // two modules stacked along N
#define KCH     25      // ceil(784/32)

// ---- workspace layout ----
// halves region (f16), element offsets:
#define W1P_OFF_H   0                        // 25 chunks * 3 tiles * 512 = 38400
#define W2P_OFF_H   38400                    // 3 * 512 = 1536
#define W3P_OFF_H   (38400 + 1536)           // 1536
#define WLP_OFF_H   (38400 + 1536 + 1536)    // 512
#define HALVES_TOTAL (38400 + 1536 + 1536 + 512)   // 41984 halves = 83968 bytes (16B aligned)
// floats region (f32), element offsets (base = ws + HALVES_TOTAL*2 bytes):
#define B1P_OFF_F   0
#define B2P_OFF_F   48
#define B3P_OFF_F   96
#define BLP_OFF_F   144
#define FLOATS_TOTAL 160

// K index inside a 32-wide chunk for half h (0..15) given lane-half hi (0/1),
// per the CDNA5 16-bit A/B fragment layout:
// lanes 0-15: K = {0..7, 16..23}; lanes 16-31: K = {8..15, 24..31}
__device__ __forceinline__ int kmap(int h, int hi) {
    return (h < 8) ? (hi * 8 + h) : (hi * 8 + 16 + (h - 8));
}

// sorted(set(row))[:2] semantics from the reference; second index -1 if only one unique
__device__ __forceinline__ void decode_pathway(const long long* pw, int l, int& ia, int& ib) {
    int a0 = (int)pw[l * 3 + 0], a1 = (int)pw[l * 3 + 1], a2 = (int)pw[l * 3 + 2];
    int t;
    if (a0 > a1) { t = a0; a0 = a1; a1 = t; }
    if (a1 > a2) { t = a1; a1 = a2; a2 = t; }
    if (a0 > a1) { t = a0; a0 = a1; a1 = t; }
    ia = a0;
    ib = (a1 != a0) ? a1 : ((a2 != a0) ? a2 : -1);
}

// ---------------- prep: pack weights into WMMA B-fragment order (f16) ----------------
__global__ void pathnet_prep(const float* __restrict__ W1, const float* __restrict__ b1,
                             const float* __restrict__ W2, const float* __restrict__ b2,
                             const float* __restrict__ W3, const float* __restrict__ b3,
                             const float* __restrict__ Wl, const float* __restrict__ bl,
                             const long long* __restrict__ pw,
                             _Float16* __restrict__ wsH, float* __restrict__ wsF)
{
    int i1a, i1b, i2a, i2b, i3a, i3b;
    decode_pathway(pw, 0, i1a, i1b);
    decode_pathway(pw, 1, i2a, i2b);
    decode_pathway(pw, 2, i3a, i3b);

    const int total = HALVES_TOTAL + FLOATS_TOTAL;
    for (int i = blockIdx.x * blockDim.x + threadIdx.x; i < total;
         i += gridDim.x * blockDim.x) {
        if (i < HALVES_TOTAL) {
            float v = 0.f;
            if (i < W2P_OFF_H) {                       // layer-1 weights, K=784 -> 25 chunks
                int e = i;
                int ct = e >> 9, w = e & 511;
                int chunk = ct / 3, tile = ct % 3;
                int lane = w >> 4, h = w & 15;
                int hi = lane >> 4, n = lane & 15;
                int kg = chunk * 32 + kmap(h, hi);
                int ng = tile * 16 + n;
                if (kg < IN_DIM && ng < NSTACK) {
                    int mod = (ng < HW) ? i1a : i1b;
                    int r   = (ng < HW) ? ng  : ng - HW;
                    if (mod >= 0) v = W1[(size_t)mod * HW * IN_DIM + (size_t)r * IN_DIM + kg];
                }
            } else if (i < WLP_OFF_H) {                // layer-2/3 weights, K=20 (one chunk)
                int layer3 = (i >= W3P_OFF_H);
                int e = i - (layer3 ? W3P_OFF_H : W2P_OFF_H);
                int tile = e >> 9, w = e & 511;
                int lane = w >> 4, h = w & 15;
                int hi = lane >> 4, n = lane & 15;
                int kg = kmap(h, hi);
                int ng = tile * 16 + n;
                if (kg < HW && ng < NSTACK) {
                    int mod = (ng < HW) ? (layer3 ? i3a : i2a) : (layer3 ? i3b : i2b);
                    int r   = (ng < HW) ? ng : ng - HW;
                    if (mod >= 0) {
                        const float* W = layer3 ? W3 : W2;
                        v = W[(size_t)mod * HW * HW + (size_t)r * HW + kg];
                    }
                }
            } else {                                   // head weights Wl [10 x 20]
                int e = i - WLP_OFF_H;
                int lane = e >> 4, h = e & 15;
                int hi = lane >> 4, n = lane & 15;
                int kg = kmap(h, hi);
                if (kg < HW && n < NCLS) v = Wl[n * HW + kg];
            }
            wsH[i] = (_Float16)v;
        } else {
            int j = i - HALVES_TOTAL;
            float v;
            if (j < 144) {                             // packed biases for layers 1..3
                int layer = j / 48;                    // 0,1,2
                int ng = j % 48;
                v = -3.0e38f;                          // absent module / pad -> relu kills it
                if (ng < NSTACK) {
                    int mod, r = (ng < HW) ? ng : ng - HW;
                    const float* b;
                    if (layer == 0)      { mod = (ng < HW) ? i1a : i1b; b = b1; }
                    else if (layer == 1) { mod = (ng < HW) ? i2a : i2b; b = b2; }
                    else                 { mod = (ng < HW) ? i3a : i3b; b = b3; }
                    if (mod >= 0) v = b[mod * HW + r];
                }
            } else {                                   // head bias
                int n = j - 144;
                v = (n < NCLS) ? bl[n] : 0.f;
            }
            wsF[j] = v;
        }
    }
}

// ---------------- fused forward ----------------
__device__ __forceinline__ v8f wmma_f16(v16h a, v16h b, v8f c) {
    return __builtin_amdgcn_wmma_f32_16x16x32_f16(false, a, false, b, (short)0, c,
                                                  false, false);
}

// Build an A fragment from a row of streaming f32 x with non-temporal loads.
__device__ __forceinline__ v16h load_a_x(const float* __restrict__ xr, int kb, bool full) {
    const v4f* p = (const v4f*)(xr + kb);
    v4f q0 = __builtin_nontemporal_load(p + 0);
    v4f q1 = __builtin_nontemporal_load(p + 1);
    v4f q2 = {0.f, 0.f, 0.f, 0.f}, q3 = {0.f, 0.f, 0.f, 0.f};
    if (full) {                                       // K tail 784..799 zero-padded
        q2 = __builtin_nontemporal_load(p + 4);
        q3 = __builtin_nontemporal_load(p + 5);
    }
    v16h a;
    a[0]  = (_Float16)q0[0]; a[1]  = (_Float16)q0[1]; a[2]  = (_Float16)q0[2]; a[3]  = (_Float16)q0[3];
    a[4]  = (_Float16)q1[0]; a[5]  = (_Float16)q1[1]; a[6]  = (_Float16)q1[2]; a[7]  = (_Float16)q1[3];
    a[8]  = (_Float16)q2[0]; a[9]  = (_Float16)q2[1]; a[10] = (_Float16)q2[2]; a[11] = (_Float16)q2[3];
    a[12] = (_Float16)q3[0]; a[13] = (_Float16)q3[1]; a[14] = (_Float16)q3[2]; a[15] = (_Float16)q3[3];
    return a;
}

// relu + bias + pair-sum of one row of the staged [.,48] C buffer -> next A fragment
__device__ __forceinline__ v16h relu_pack(const float* crow, const float* __restrict__ bias,
                                          int hi) {
    v16h an;
#pragma unroll
    for (int h = 0; h < 16; ++h) {
        int k = kmap(h, hi);
        float v = 0.f;
        if (k < HW) {
            float u0 = crow[k]      + bias[k];
            float u1 = crow[HW + k] + bias[HW + k];
            v = fmaxf(u0, 0.f) + fmaxf(u1, 0.f);
        }
        an[h] = (_Float16)v;
    }
    return an;
}

// one small expert layer over two 16-row tiles (K=20 pad 32, N=40 pad 48): 6 WMMAs,
// stage C through LDS, produce both next-layer A fragments.
__device__ __forceinline__ void expert_small2(v16h& aA, v16h& aB,
                                              const _Float16* __restrict__ wpack,
                                              const float* __restrict__ bias,
                                              float cb[32][48], int lane)
{
    const int hi = lane >> 4, lo = lane & 15, mb = hi * 8;
    v16h b0 = *(const v16h*)(wpack + 0 * 512 + lane * 16);
    v16h b1 = *(const v16h*)(wpack + 1 * 512 + lane * 16);
    v16h b2 = *(const v16h*)(wpack + 2 * 512 + lane * 16);
    v8f c00 = {}, c01 = {}, c02 = {}, c10 = {}, c11 = {}, c12 = {};
    c00 = wmma_f16(aA, b0, c00);
    c10 = wmma_f16(aB, b0, c10);
    c01 = wmma_f16(aA, b1, c01);
    c11 = wmma_f16(aB, b1, c11);
    c02 = wmma_f16(aA, b2, c02);
    c12 = wmma_f16(aB, b2, c12);
#pragma unroll
    for (int r = 0; r < 8; ++r) {
        cb[mb + r][ 0 + lo] = c00[r];
        cb[mb + r][16 + lo] = c01[r];
        cb[mb + r][32 + lo] = c02[r];
        cb[16 + mb + r][ 0 + lo] = c10[r];
        cb[16 + mb + r][16 + lo] = c11[r];
        cb[16 + mb + r][32 + lo] = c12[r];
    }
    __syncthreads();
    aA = relu_pack(cb[lo],      bias, hi);
    aB = relu_pack(cb[16 + lo], bias, hi);
    __syncthreads();
}

__global__ __launch_bounds__(256)
void pathnet_main(const float* __restrict__ x, const _Float16* __restrict__ wsH,
                  const float* __restrict__ wsF, float* __restrict__ out)
{
    __shared__ float cbuf[8][32][48];
    const int lane = threadIdx.x & 31;
    const int wave = threadIdx.x >> 5;
    const int hi = lane >> 4;
    const int lo = lane & 15;
    const int row0 = blockIdx.x * 256 + wave * 32;    // 8 waves x 32 rows
    const float* xrA = x + (size_t)(row0 + lo) * IN_DIM;
    const float* xrB = x + (size_t)(row0 + 16 + lo) * IN_DIM;

    // ----- layer 1: [32 x 784] x [784 x 48], 25 K-chunks, B shared by both row tiles -----
    v8f c00 = {}, c01 = {}, c02 = {}, c10 = {}, c11 = {}, c12 = {};
    for (int c = 0; c < KCH; ++c) {
        const int kb = c * 32 + hi * 8;
        const bool full = (c < KCH - 1);
        v16h aA = load_a_x(xrA, kb, full);
        v16h aB = load_a_x(xrB, kb, full);
        const _Float16* wp = wsH + (size_t)(c * 3) * 512;
        v16h b0 = *(const v16h*)(wp + 0 * 512 + lane * 16);
        v16h b1 = *(const v16h*)(wp + 1 * 512 + lane * 16);
        v16h b2 = *(const v16h*)(wp + 2 * 512 + lane * 16);
        c00 = wmma_f16(aA, b0, c00);
        c10 = wmma_f16(aB, b0, c10);
        c01 = wmma_f16(aA, b1, c01);
        c11 = wmma_f16(aB, b1, c11);
        c02 = wmma_f16(aA, b2, c02);
        c12 = wmma_f16(aB, b2, c12);
    }
    {
        const int mb = hi * 8;
#pragma unroll
        for (int r = 0; r < 8; ++r) {
            cbuf[wave][mb + r][ 0 + lo] = c00[r];
            cbuf[wave][mb + r][16 + lo] = c01[r];
            cbuf[wave][mb + r][32 + lo] = c02[r];
            cbuf[wave][16 + mb + r][ 0 + lo] = c10[r];
            cbuf[wave][16 + mb + r][16 + lo] = c11[r];
            cbuf[wave][16 + mb + r][32 + lo] = c12[r];
        }
    }
    __syncthreads();
    v16h aA = relu_pack(cbuf[wave][lo],      wsF + B1P_OFF_F, hi);
    v16h aB = relu_pack(cbuf[wave][16 + lo], wsF + B1P_OFF_F, hi);
    __syncthreads();

    // ----- layers 2 and 3 -----
    expert_small2(aA, aB, wsH + W2P_OFF_H, wsF + B2P_OFF_F, cbuf[wave], lane);
    expert_small2(aA, aB, wsH + W3P_OFF_H, wsF + B3P_OFF_F, cbuf[wave], lane);

    // ----- head: [32 x 20] x [20 x 10] (padded 32x16), 2 WMMAs -----
    v8f cA = {}, cB = {};
    {
        v16h bw = *(const v16h*)(wsH + WLP_OFF_H + lane * 16);
        cA = wmma_f16(aA, bw, cA);
        cB = wmma_f16(aB, bw, cB);
    }
    if (lo < NCLS) {
        const float blv = wsF[BLP_OFF_F + lo];
#pragma unroll
        for (int r = 0; r < 8; ++r) {
            int rowA = row0 + hi * 8 + r;
            int rowB = row0 + 16 + hi * 8 + r;
            out[(size_t)rowA * NCLS + lo] = cA[r] + blv;
            out[(size_t)rowB * NCLS + lo] = cB[r] + blv;
        }
    }
}

extern "C" void kernel_launch(void* const* d_in, const int* in_sizes, int n_in,
                              void* d_out, int out_size, void* d_ws, size_t ws_size,
                              hipStream_t stream) {
    const float* x       = (const float*)d_in[0];
    const float* W1      = (const float*)d_in[1];
    const float* b1      = (const float*)d_in[2];
    const float* W2      = (const float*)d_in[3];
    const float* b2      = (const float*)d_in[4];
    const float* W3      = (const float*)d_in[5];
    const float* b3      = (const float*)d_in[6];
    const float* Wl      = (const float*)d_in[7];
    const float* bl      = (const float*)d_in[8];
    const long long* pw  = (const long long*)d_in[9];

    _Float16* wsH = (_Float16*)d_ws;
    float*    wsF = (float*)((char*)d_ws + (size_t)HALVES_TOTAL * sizeof(_Float16));

    pathnet_prep<<<64, 256, 0, stream>>>(W1, b1, W2, b2, W3, b3, Wl, bl, pw, wsH, wsF);
    pathnet_main<<<BATCH / 256, 256, 0, stream>>>(x, wsH, wsF, (float*)d_out);
}
